// LocalStyleAdaptor_30760555774513
// MI455X (gfx1250) — matrix operand order repaired and checked by reading, not compiled, not run on hardware
//
#include <hip/hip_runtime.h>
#include <hip/hip_bf16.h>
#include <math.h>

// ---------------- problem constants ----------------
#define BB   16
#define TT   4000
#define BTR  (BB*TT)        // 64000 rows
#define CCH  80             // mel channels
#define CPAD 96             // 80 padded to 96 (3 x 32)
#define HD   256            // VQ dim
#define MCODE 64            // num codes
#define NRES 10

typedef __attribute__((ext_vector_type(16))) __bf16 v16bf;
typedef __attribute__((ext_vector_type(8)))  float  v8f;
typedef __attribute__((ext_vector_type(4)))  unsigned int u32x4;
typedef __attribute__((ext_vector_type(8)))  int i32x8;
typedef __attribute__((ext_vector_type(4)))  int i32x4;

union Frag16 { uint4 u[2]; v16bf v; };

// =====================================================================
// Conv-as-GEMM with WMMA bf16 (fp32 accumulate), TDM-staged activations.
// x      : [B*T, CIN_PAD] bf16 (channel-padded with zeros)
// wpack  : B fragments, [coutTile][k][cinChunk][lane][16] bf16
// Block = 80 output rows (5 M-tiles) x all COUT tiles; 8 waves loop tiles.
// Activation rows [tloc-PAD, tloc+79+PAD] are DMA'd to LDS by the Tensor
// Data Mover (wave 0 issues the descriptor; bottom-of-batch halo handled
// by TDM OOB-zero semantics, top halo of the first block zero-filled).
// MODE 0 : y_f32 = acc + bias
// MODE 1 : y_bf16 = GELU((acc+bias)*scale)
// MODE 2 : v=(acc+bias)*nonpad ; y_f32=v ; y_bf16=v
// =====================================================================
template<int KTAPS, int CIN_PAD, int MODE, int COUT>
__global__ __launch_bounds__(256)
void conv_wmma(const __hip_bfloat16* __restrict__ x,
               const __hip_bfloat16* __restrict__ wpack,
               const float* __restrict__ bias,
               float* __restrict__ yf,
               __hip_bfloat16* __restrict__ yb,
               const float* __restrict__ nonpad,
               float scale)
{
    constexpr int PAD  = (KTAPS - 1) / 2;
    constexpr int ROWS = 80;                 // output rows per block (5 M-tiles)
    constexpr int R    = ROWS + KTAPS - 1;   // staged rows incl. halo
    constexpr int CH   = CIN_PAD / 32;
    constexpr int CT   = COUT / 16;
    constexpr int NT   = ROWS / 16;
    __shared__ __align__(16) __hip_bfloat16 sx[R * CIN_PAD];

    const int b    = blockIdx.z;
    const int tloc = blockIdx.x * ROWS;
    const int lead = (tloc < PAD) ? (PAD - tloc) : 0;  // top-halo rows to zero

    // zero top halo rows (only first block of each batch, tiny)
    for (int i = threadIdx.x; i < lead * CIN_PAD / 8; i += 256)
        *(uint4*)&sx[i * 8] = make_uint4(0u, 0u, 0u, 0u);

    if (threadIdx.x < 32) {
        // ---- Tensor Data Mover: 2D tile (CIN_PAD x (R-lead) rows) -> LDS ----
        const long long gRow = (long long)b * TT + tloc - PAD + lead;
        const unsigned long long gaddr =
            (unsigned long long)(uintptr_t)(x + gRow * (long long)CIN_PAD);
        const unsigned ldsAddr =
            (unsigned)(uintptr_t)(&sx[0]) + (unsigned)(lead * CIN_PAD * 2);
        const unsigned validRows = (unsigned)(TT - (tloc - PAD + lead)); // OOB rows read 0
        const unsigned tileRows  = (unsigned)(R - lead);

        // D# group 0: count=1 | lds_addr | global_addr[56:0] | type=2
        u32x4 g0 = { 1u,
                     ldsAddr,
                     (unsigned)(gaddr & 0xffffffffull),
                     (unsigned)((gaddr >> 32) & 0x1ffffffull) | 0x80000000u };
        // D# group 1: data_size=2B, tensor_dim0/1, tile_dim0/1, dim0 stride
        const unsigned long long s0 = (unsigned long long)CIN_PAD;
        i32x8 g1 = { (int)0x00010000u,                                   // data_size=1 (2B)
                     (int)(((unsigned)CIN_PAD & 0xffffu) << 16),         // tensor_dim0 lo
                     (int)(((unsigned)CIN_PAD >> 16) | ((validRows & 0xffffu) << 16)),
                     (int)((validRows >> 16) | ((unsigned)CIN_PAD << 16)), // tile_dim0
                     (int)(tileRows),                                    // tile_dim1, tile_dim2=0
                     (int)(s0 & 0xffffffffull),                          // dim0 stride lo
                     (int)(s0 >> 32),                                    // stride hi, stride1 lo=0
                     0 };
        i32x4 gz4 = { 0, 0, 0, 0 };
        i32x8 gz8 = { 0, 0, 0, 0, 0, 0, 0, 0 };
        __builtin_amdgcn_tensor_load_to_lds(g0, g1, gz4, gz4, gz8, 0);
        __builtin_amdgcn_s_wait_tensorcnt(0);
    }
    __syncthreads();

    const int wv    = threadIdx.x >> 5;
    const int ln    = threadIdx.x & 31;
    const int lrow  = ln & 15;
    const int hiSel = ln >> 4;
    const int col   = ln & 15;

    for (int wt = wv; wt < NT * CT; wt += 8) {
        const int tt = wt / CT;
        const int ct = wt - tt * CT;

        v8f acc = {0.f,0.f,0.f,0.f,0.f,0.f,0.f,0.f};
        #pragma unroll
        for (int k = 0; k < KTAPS; ++k) {
            #pragma unroll
            for (int c = 0; c < CH; ++c) {
                Frag16 a, bw;
                // A fragment 16x32: lanes 0-15 K{0..7,16..23}, lanes 16-31 K{8..15,24..31}
                const int ab = (tt * 16 + lrow + k) * CIN_PAD + c * 32 + hiSel * 8;
                a.u[0] = *(const uint4*)&sx[ab];
                a.u[1] = *(const uint4*)&sx[ab + 16];
                // B fragment pre-packed in lane order -> 2 contiguous b128 loads
                const __hip_bfloat16* wp =
                    wpack + ((size_t)((ct * KTAPS + k) * CH + c) * 32 + ln) * 16;
                bw.u[0] = *(const uint4*)(wp);
                bw.u[1] = *(const uint4*)(wp + 8);
                acc = __builtin_amdgcn_wmma_f32_16x16x32_bf16(
                          false, a.v, false, bw.v, (short)0, acc, false, false);
            }
        }

        const int n  = ct * 16 + col;
        const float bs = bias ? bias[n] : 0.0f;
        #pragma unroll
        for (int r = 0; r < 8; ++r) {
            const int m = tt * 16 + hiSel * 8 + r;    // C/D layout: VGPR r -> M=r (+8 hi half)
            const size_t grow = (size_t)b * TT + tloc + m;
            float v = acc[r] + bs;
            if (MODE == 1) {
                v *= scale;
                v = 0.5f * v * (1.0f + erff(v * 0.70710678118654752f));
                yb[grow * (size_t)COUT + n] = __float2bfloat16(v);
            } else if (MODE == 2) {
                v *= nonpad[grow];
                yf[grow * (size_t)COUT + n] = v;
                yb[grow * (size_t)COUT + n] = __float2bfloat16(v);
            } else {
                yf[grow * (size_t)COUT + n] = v;
            }
        }
    }
}

// =====================================================================
// Weight packer: w[Cout][Cin][K] f32 -> B fragments bf16
// =====================================================================
__global__ void pack_w(const float* __restrict__ w, __hip_bfloat16* __restrict__ out,
                       int Cout, int Cin, int K, int CinPad)
{
    int tid = blockIdx.x * blockDim.x + threadIdx.x;
    int chunks = CinPad / 32;
    int coutTiles = (Cout + 15) / 16;
    int nfrag = coutTiles * K * chunks;
    if (tid >= nfrag * 32) return;
    int f = tid >> 5, ln = tid & 31;
    int c = f % chunks;
    int k = (f / chunks) % K;
    int ct = f / (chunks * K);
    int col = ln & 15, cout = ct * 16 + col;
    int kb = (ln < 16) ? 0 : 8;
    __hip_bfloat16* o = out + (size_t)tid * 16;
    for (int h = 0; h < 16; ++h) {
        int cin = c * 32 + kb + h + ((h >= 8) ? 8 : 0);
        float v = (cout < Cout && cin < Cin) ? w[((size_t)cout * Cin + cin) * K + k] : 0.0f;
        o[h] = __float2bfloat16(v);
    }
}

// ---------------- small elementwise / reduction kernels ----------------
__global__ void prep_kernel(const float* __restrict__ ref, float* __restrict__ mask,
                            float* __restrict__ hf, __hip_bfloat16* __restrict__ hbf,
                            float* __restrict__ outw)
{
    int row = blockIdx.x * blockDim.x + threadIdx.x;
    if (row >= BTR) return;
    const float* r = ref + (size_t)row * CCH;
    mask[row] = (r[0] != 0.0f) ? 1.0f : 0.0f;
    for (int c = 0; c < CCH; ++c) {
        float v = r[c];
        hf[(size_t)row * CCH + c] = v;
        hbf[(size_t)row * CPAD + c] = __float2bfloat16(v);
        outw[(size_t)row * CCH + c] = 0.0f;
    }
    for (int c = CCH; c < CPAD; ++c) hbf[(size_t)row * CPAD + c] = __float2bfloat16(0.0f);
}

__global__ void gate_kernel(const float* __restrict__ xi, __hip_bfloat16* __restrict__ actbf)
{
    int tid = blockIdx.x * blockDim.x + threadIdx.x;
    if (tid >= BTR * CPAD) return;
    int row = tid / CPAD, c = tid % CPAD;
    float a = 0.0f;
    if (c < CCH) {
        float t = tanhf(xi[(size_t)row * 160 + c]);
        float s = 1.0f / (1.0f + expf(-xi[(size_t)row * 160 + CCH + c]));
        a = t * s;
    }
    actbf[(size_t)row * CPAD + c] = __float2bfloat16(a);
}

template<bool LAST>
__global__ void resid_wn_kernel(const float* __restrict__ rs, const float* __restrict__ mask,
                                float* __restrict__ hf, __hip_bfloat16* __restrict__ hbf,
                                float* __restrict__ outw)
{
    int tid = blockIdx.x * blockDim.x + threadIdx.x;
    if (tid >= BTR * CCH) return;
    int row = tid / CCH, c = tid % CCH;
    if (!LAST) {
        float hn = (hf[tid] + rs[(size_t)row * 160 + c]) * mask[row];
        hf[tid] = hn;
        hbf[(size_t)row * CPAD + c] = __float2bfloat16(hn);
        outw[tid] += rs[(size_t)row * 160 + CCH + c];
    } else {
        outw[tid] += rs[(size_t)row * CCH + c];
    }
}

__global__ void finish_wn_kernel(const float* __restrict__ outw, const float* __restrict__ mask,
                                 float* __restrict__ xe, float* __restrict__ nonpad)
{
    int row = blockIdx.x * blockDim.x + threadIdx.x;
    if (row >= BTR) return;
    float m = mask[row], sa = 0.0f;
    for (int c = 0; c < CCH; ++c) {
        float v = outw[(size_t)row * CCH + c] * m;
        xe[(size_t)row * CCH + c] = v;
        sa += fabsf(v);
    }
    nonpad[row] = (sa > 0.0f) ? 1.0f : 0.0f;
}

// LayerNorm over 80 channels; one wave per row, shfl_xor reductions
__global__ __launch_bounds__(256)
void ln_kernel(const float* __restrict__ xe, const float* __restrict__ gw,
               const float* __restrict__ gb, const float* __restrict__ nonpad,
               __hip_bfloat16* __restrict__ out96, int applyNp)
{
    int row = blockIdx.x * 8 + (threadIdx.x >> 5);
    int ln  = threadIdx.x & 31;
    const float* xr = xe + (size_t)row * CCH;
    float v0 = (ln      < CCH) ? xr[ln]      : 0.0f;
    float v1 = (ln + 32 < CCH) ? xr[ln + 32] : 0.0f;
    float v2 = (ln + 64 < CCH) ? xr[ln + 64] : 0.0f;
    float s = v0 + v1 + v2, sq = v0 * v0 + v1 * v1 + v2 * v2;
    for (int m = 16; m >= 1; m >>= 1) { s += __shfl_xor(s, m, 32); sq += __shfl_xor(sq, m, 32); }
    float mu = s * (1.0f / 80.0f);
    float var = sq * (1.0f / 80.0f) - mu * mu;
    float rsq = rsqrtf(var + 1e-5f);
    float np = applyNp ? nonpad[row] : 1.0f;
    __hip_bfloat16* o = out96 + (size_t)row * CPAD;
    o[ln]      = __float2bfloat16(((v0 - mu) * rsq * gw[ln]      + gb[ln])      * np);
    o[ln + 32] = __float2bfloat16((ln + 32 < CCH) ? ((v1 - mu) * rsq * gw[ln + 32] + gb[ln + 32]) * np : 0.0f);
    if (ln < 16)
        o[ln + 64] = __float2bfloat16(((v2 - mu) * rsq * gw[ln + 64] + gb[ln + 64]) * np);
    else
        o[ln + 64] = __float2bfloat16(0.0f);   // channels 80..95 zero pad
}

__global__ void enc_resid_kernel(float* __restrict__ xe, const float* __restrict__ hh,
                                 const float* __restrict__ nonpad)
{
    int tid = blockIdx.x * blockDim.x + threadIdx.x;
    if (tid >= BTR * CCH) return;
    int row = tid / CCH;
    xe[tid] = (xe[tid] + hh[tid]) * nonpad[row];
}

__global__ void e2_kernel(const float* __restrict__ emb, float* __restrict__ e2)
{
    int m = threadIdx.x;
    if (m >= MCODE) return;
    float s = 0.0f;
    for (int h = 0; h < HD; ++h) { float v = emb[(size_t)m * HD + h]; s += v * v; }
    e2[m] = s;
}

__global__ void init_kernel(float* __restrict__ counts, float* __restrict__ scal)
{
    int t = threadIdx.x;
    if (t < MCODE) counts[t] = 0.0f;
    if (t < 2)     scal[t]   = 0.0f;
}

__global__ void vq_argmin_kernel(const float* __restrict__ dots, const float* __restrict__ e2,
                                 int* __restrict__ idx, float* __restrict__ counts)
{
    int row = blockIdx.x * blockDim.x + threadIdx.x;
    if (row >= BTR) return;
    const float* d = dots + (size_t)row * MCODE;
    float best = 1e30f; int bi = 0;
    for (int m = 0; m < MCODE; ++m) {
        float dist = e2[m] - 2.0f * d[m];
        if (dist < best) { best = dist; bi = m; }
    }
    idx[row] = bi;
    atomicAdd(&counts[bi], 1.0f);
}

// z = q (straight-through); accumulate commit-loss numerator/denominator
__global__ __launch_bounds__(256)
void vq_out_kernel(float* __restrict__ z /* holds pros, overwritten */,
                   const float* __restrict__ emb, const int* __restrict__ idx,
                   float* __restrict__ scal)
{
    __shared__ float s1[256], s2[256];
    int row = blockIdx.x, t = threadIdx.x;
    float p = z[(size_t)row * HD + t];
    float q = emb[(size_t)idx[row] * HD + t];
    z[(size_t)row * HD + t] = q;
    float d = p - q;
    s1[t] = d * d;
    s2[t] = fabsf(p);
    __syncthreads();
    for (int s = 128; s > 0; s >>= 1) {
        if (t < s) { s1[t] += s1[t + s]; s2[t] += s2[t + s]; }
        __syncthreads();
    }
    if (t == 0) {
        float np = (s2[0] > 0.0f) ? 1.0f : 0.0f;
        atomicAdd(&scal[0], np * s1[0] * (1.0f / 256.0f));
        atomicAdd(&scal[1], np);
    }
}

__global__ void finalize_kernel(const float* __restrict__ counts, const float* __restrict__ scal,
                                float* __restrict__ out)
{
    __shared__ float sh[64];
    int t = threadIdx.x;
    float p = counts[t] * (1.0f / (float)BTR);
    sh[t] = p * logf(p + 1e-10f);
    __syncthreads();
    for (int s = 32; s > 0; s >>= 1) { if (t < s) sh[t] += sh[t + s]; __syncthreads(); }
    if (t == 0) {
        out[0] = 0.25f * scal[0] / fmaxf(scal[1], 1.0f);  // commit loss
        out[1] = expf(-sh[0]);                            // perplexity
    }
}

// =====================================================================
extern "C" void kernel_launch(void* const* d_in, const int* in_sizes, int n_in,
                              void* d_out, int out_size, void* d_ws, size_t ws_size,
                              hipStream_t stream)
{
    (void)in_sizes; (void)n_in; (void)out_size; (void)ws_size;
    const float* ref       = (const float*)d_in[0];
    const float* wn_in_w   = (const float*)d_in[1];   // [4,160,80,3]
    const float* wn_in_b   = (const float*)d_in[2];   // [4,160]
    const float* wn_rs_w[4] = {(const float*)d_in[3], (const float*)d_in[4],
                               (const float*)d_in[5], (const float*)d_in[6]};
    const float* wn_rs_b[4] = {(const float*)d_in[7], (const float*)d_in[8],
                               (const float*)d_in[9], (const float*)d_in[10]};
    const float* enc_ln_w  = (const float*)d_in[11];  // [10,80]
    const float* enc_ln_b  = (const float*)d_in[12];
    const float* enc_w1    = (const float*)d_in[13];  // [10,160,80,5]
    const float* enc_b1    = (const float*)d_in[14];  // [10,160]
    const float* enc_w2    = (const float*)d_in[15];  // [10,80,160,1]
    const float* enc_b2    = (const float*)d_in[16];  // [10,80]
    const float* last_ln_w = (const float*)d_in[17];
    const float* last_ln_b = (const float*)d_in[18];
    const float* post_w    = (const float*)d_in[19];  // [256,80,3]
    const float* post_b    = (const float*)d_in[20];  // [256]
    const float* emb       = (const float*)d_in[21];  // [64,256]

    // ---- workspace arena ----
    char* ws = (char*)d_ws;
    size_t off = 0;
    auto arena = [&](size_t bytes) { size_t o = off; off = (off + bytes + 255) & ~(size_t)255; return (void*)(ws + o); };
    float*           mask   = (float*)arena((size_t)BTR * 4);
    float*           nonpad = (float*)arena((size_t)BTR * 4);
    __hip_bfloat16*  hbf    = (__hip_bfloat16*)arena((size_t)BTR * CPAD * 2);   // h / ln bf16
    float*           hf     = (float*)arena((size_t)BTR * CCH * 4);             // h / xe master
    float*           outw   = (float*)arena((size_t)BTR * CCH * 4);             // WN skip accum
    __hip_bfloat16*  actbf  = (__hip_bfloat16*)arena((size_t)BTR * 160 * 2);    // gated/GELU acts
    float*           f32a   = (float*)arena((size_t)BTR * 160 * 4);             // xi/rs/hh/dots
    __hip_bfloat16*  prosbf = (__hip_bfloat16*)arena((size_t)BTR * HD * 2);
    int*             idxb   = (int*)arena((size_t)BTR * 4);
    float*           counts = (float*)arena(MCODE * 4);
    float*           e2b    = (float*)arena(MCODE * 4);
    float*           scal   = (float*)arena(8 * 4);

    auto packBytes = [](int Cout, int K, int CinPad) {
        return (size_t)((Cout + 15) / 16) * K * (CinPad / 32) * 32 * 16 * 2;
    };
    auto launch_pack = [&](const float* w, __hip_bfloat16* dst, int Cout, int Cin, int K, int CinPad) {
        int nthr = ((Cout + 15) / 16) * K * (CinPad / 32) * 32;
        pack_w<<<(nthr + 255) / 256, 256, 0, stream>>>(w, dst, Cout, Cin, K, CinPad);
    };

    __hip_bfloat16* pk_wnin[4], *pk_wnrs[4], *pk_enc1[NRES], *pk_enc2[NRES], *pk_post, *pk_emb;
    for (int i = 0; i < 4; ++i)    pk_wnin[i] = (__hip_bfloat16*)arena(packBytes(160, 3, CPAD));
    for (int i = 0; i < 4; ++i)    pk_wnrs[i] = (__hip_bfloat16*)arena(packBytes(160, 1, CPAD));
    for (int i = 0; i < NRES; ++i) pk_enc1[i] = (__hip_bfloat16*)arena(packBytes(160, 5, CPAD));
    for (int i = 0; i < NRES; ++i) pk_enc2[i] = (__hip_bfloat16*)arena(packBytes(80, 1, 160));
    pk_post = (__hip_bfloat16*)arena(packBytes(256, 3, CPAD));
    pk_emb  = (__hip_bfloat16*)arena(packBytes(MCODE, 1, HD));

    // ---- pack all weights (bf16 fragment order) ----
    for (int i = 0; i < 4; ++i)    launch_pack(wn_in_w + (size_t)i * 160 * 80 * 3, pk_wnin[i], 160, 80, 3, CPAD);
    for (int i = 0; i < 3; ++i)    launch_pack(wn_rs_w[i], pk_wnrs[i], 160, 80, 1, CPAD);
    launch_pack(wn_rs_w[3], pk_wnrs[3], 80, 80, 1, CPAD);
    for (int i = 0; i < NRES; ++i) launch_pack(enc_w1 + (size_t)i * 160 * 80 * 5, pk_enc1[i], 160, 80, 5, CPAD);
    for (int i = 0; i < NRES; ++i) launch_pack(enc_w2 + (size_t)i * 80 * 160,     pk_enc2[i], 80, 160, 1, 160);
    launch_pack(post_w, pk_post, 256, 80, 3, CPAD);
    launch_pack(emb,    pk_emb, MCODE, HD, 1, HD);
    e2_kernel<<<1, 64, 0, stream>>>(emb, e2b);
    init_kernel<<<1, 128, 0, stream>>>(counts, scal);

    const dim3 blk(256);
    const dim3 cgrid(TT / 80, 1, BB);    // 50 row-chunks x 16 batches

    // ---- prep: mask, h0 = x, out = 0 ----
    prep_kernel<<<(BTR + 255) / 256, blk, 0, stream>>>(ref, mask, hf, hbf, outw);

    // ---- WaveNet (4 layers) ----
    for (int i = 0; i < 4; ++i) {
        conv_wmma<3, CPAD, 0, 160><<<cgrid, blk, 0, stream>>>(
            hbf, pk_wnin[i], wn_in_b + (size_t)i * 160, f32a, nullptr, nullptr, 1.0f);
        gate_kernel<<<((size_t)BTR * CPAD + 255) / 256, blk, 0, stream>>>(f32a, actbf);
        if (i < 3)
            conv_wmma<1, CPAD, 0, 160><<<cgrid, blk, 0, stream>>>(
                actbf, pk_wnrs[i], wn_rs_b[i], f32a, nullptr, nullptr, 1.0f);
        else
            conv_wmma<1, CPAD, 0, 80><<<cgrid, blk, 0, stream>>>(
                actbf, pk_wnrs[i], wn_rs_b[i], f32a, nullptr, nullptr, 1.0f);
        if (i < 3)
            resid_wn_kernel<false><<<((size_t)BTR * CCH + 255) / 256, blk, 0, stream>>>(f32a, mask, hf, hbf, outw);
        else
            resid_wn_kernel<true><<<((size_t)BTR * CCH + 255) / 256, blk, 0, stream>>>(f32a, mask, hf, hbf, outw);
    }
    finish_wn_kernel<<<(BTR + 255) / 256, blk, 0, stream>>>(outw, mask, hf, nonpad);

    // ---- ConvBlocks encoder (10 blocks) ----
    const float gscale = 0.44721359549995794f;   // 5^-0.5
    for (int i = 0; i < NRES; ++i) {
        ln_kernel<<<BTR / 8, blk, 0, stream>>>(hf, enc_ln_w + (size_t)i * CCH,
                                               enc_ln_b + (size_t)i * CCH, nonpad, hbf, 0);
        conv_wmma<5, CPAD, 1, 160><<<cgrid, blk, 0, stream>>>(
            hbf, pk_enc1[i], enc_b1 + (size_t)i * 160, nullptr, actbf, nullptr, gscale);
        conv_wmma<1, 160, 0, 80><<<cgrid, blk, 0, stream>>>(
            actbf, pk_enc2[i], enc_b2 + (size_t)i * CCH, f32a, nullptr, nullptr, 1.0f);
        enc_resid_kernel<<<((size_t)BTR * CCH + 255) / 256, blk, 0, stream>>>(hf, f32a, nonpad);
    }

    // ---- last LN (* nonpad) + post conv -> pros (f32 into d_out, bf16 for VQ) ----
    ln_kernel<<<BTR / 8, blk, 0, stream>>>(hf, last_ln_w, last_ln_b, nonpad, hbf, 1);
    float* pros = (float*)d_out;
    conv_wmma<3, CPAD, 2, 256><<<cgrid, blk, 0, stream>>>(
        hbf, pk_post, post_b, pros, prosbf, nonpad, 1.0f);

    // ---- VQ: dot = pros . e  (WMMA GEMM, 64000x64x256), argmin, z=q, loss, ppl ----
    conv_wmma<1, HD, 0, MCODE><<<cgrid, blk, 0, stream>>>(
        prosbf, pk_emb, nullptr, f32a, nullptr, nullptr, 1.0f);
    vq_argmin_kernel<<<(BTR + 255) / 256, blk, 0, stream>>>(f32a, e2b, idxb, counts);
    vq_out_kernel<<<BTR, 256, 0, stream>>>(pros, emb, idxb, scal);
    finalize_kernel<<<1, 64, 0, stream>>>(counts, scal, pros + (size_t)BTR * HD);
}
